// TempGNN_60550448939632
// MI455X (gfx1250) — matrix operand order repaired and checked by reading, not compiled, not run on hardware
//
#include <hip/hip_runtime.h>
#include <hip/hip_bf16.h>

typedef __attribute__((ext_vector_type(16))) __bf16 v16bf;
typedef __attribute__((ext_vector_type(8)))  float  v8f;

#define N_NODES      100000
#define TIME_LEN     128
#define OUT_CH       5
#define KSZ          7
#define CSTRIDE      3
#define CONV_OUT_LEN 41            // (128-7)/3 + 1
#define FC_IN        205           // 5*41
#define K_PAD        224           // 7*32 (FC_IN padded to WMMA K multiple)
#define EMB          32
#define HID          16
#define WPB          8             // waves per block (256 threads)
#define N_TILES      (N_NODES / 16)   // 6250 (N_NODES % 16 == 0)

// ---------------------------------------------------------------------------
// Weight prep: bf16 conversions / padding into workspace.
//   fcT : [EMB][K_PAD]  = fc_w zero-padded along K         (B for FC GEMM)
//   w11 : [HID][EMB]    = g1w1 (K=32 exact)
//   w12 : [HID][32]     = g1w2 zero-padded K 16->32
//   w21 : [HID][32]     = g2w1 zero-padded K 16->32
//   w22 : [HID][32]     = g2w2 zero-padded K 16->32
// ---------------------------------------------------------------------------
__global__ void prep_weights_kernel(const float* __restrict__ fc_w,
                                    const float* __restrict__ g1w1,
                                    const float* __restrict__ g1w2,
                                    const float* __restrict__ g2w1,
                                    const float* __restrict__ g2w2,
                                    __bf16* __restrict__ fcT,
                                    __bf16* __restrict__ w11,
                                    __bf16* __restrict__ w12,
                                    __bf16* __restrict__ w21,
                                    __bf16* __restrict__ w22) {
    int tid    = blockIdx.x * blockDim.x + threadIdx.x;
    int stride = gridDim.x * blockDim.x;
    for (int i = tid; i < EMB * K_PAD; i += stride) {
        int n = i / K_PAD, k = i - n * K_PAD;
        fcT[i] = (__bf16)(k < FC_IN ? fc_w[n * FC_IN + k] : 0.0f);
    }
    for (int i = tid; i < HID * EMB; i += stride) {
        w11[i] = (__bf16)g1w1[i];
        int n = i >> 5, k = i & 31;
        float v12 = (k < HID) ? g1w2[n * HID + k] : 0.0f;
        float v21 = (k < HID) ? g2w1[n * HID + k] : 0.0f;
        float v22 = (k < HID) ? g2w2[n * HID + k] : 0.0f;
        w12[i] = (__bf16)v12;
        w21[i] = (__bf16)v21;
        w22[i] = (__bf16)v22;
    }
}

// ---------------------------------------------------------------------------
// Fused Conv1d(1->5,k7,s3)+ReLU -> FC(205->32) via WMMA.
// One wave handles a 16-node tile; conv results staged in LDS as bf16 A-tile.
// ---------------------------------------------------------------------------
__global__ void __launch_bounds__(256) conv_fc_kernel(
    const float* __restrict__ x,
    const float* __restrict__ conv_w, const float* __restrict__ conv_b,
    const __bf16* __restrict__ fcT, const float* __restrict__ fc_b,
    float* __restrict__ hout)
{
    __shared__ __bf16 sA[WPB][16][K_PAD];   // 57344 B
    __shared__ float  swt[OUT_CH * KSZ];
    __shared__ float  sbs[OUT_CH];

    int t = threadIdx.x;
    if (t < OUT_CH * KSZ) swt[t] = conv_w[t];
    if (t < OUT_CH)       sbs[t] = conv_b[t];

    int wave = t >> 5;
    int lane = t & 31;
    int tile = blockIdx.x * WPB + wave;
    int base = tile * 16;
    bool live = (tile < N_TILES);           // wave-uniform; N_NODES % 16 == 0
    __syncthreads();

    // --- conv into LDS A-tile (bf16, zero-padded K) ---
    for (int r = 0; r < 16; ++r) {
        const float* xr = x + (size_t)(base + r) * TIME_LEN;
        for (int j = lane; j < K_PAD; j += 32) {
            float v = 0.0f;
            if (live && j < FC_IN) {
                int c  = j / CONV_OUT_LEN;
                int tt = j - c * CONV_OUT_LEN;
                float acc = sbs[c];
                const float* xp = xr + tt * CSTRIDE;
                const float* wp = swt + c * KSZ;
#pragma unroll
                for (int k = 0; k < KSZ; ++k) acc += xp[k] * wp[k];
                v = acc > 0.0f ? acc : 0.0f;
            }
            sA[wave][r][j] = (__bf16)v;
        }
    }
    __syncthreads();

    // --- FC: [16 x K_PAD] @ [K_PAD x 32] with v_wmma_f32_16x16x32_bf16 ---
    int m   = lane & 15;                 // A: row M, B: col N, D: col N
    int kA  = (lane >> 4) << 3;          // unused; A also reads at kB (0 / 16)
    int kB  = (lane >> 4) << 4;          // B half-wave K base (0 / 16)
    v8f acc0 = {}, acc1 = {};
    for (int k0 = 0; k0 < K_PAD; k0 += 32) {
        v16bf a;
#pragma unroll
        for (int e = 0; e < 16; ++e)
            a[e] = sA[wave][m][k0 + kB + e];
        const __bf16* bp0 = fcT + (size_t)m * K_PAD + k0 + kB;
        const __bf16* bp1 = fcT + (size_t)(m + 16) * K_PAD + k0 + kB;
        v16bf b0, b1;
#pragma unroll
        for (int e = 0; e < 16; ++e) { b0[e] = bp0[e]; b1[e] = bp1[e]; }
        acc0 = __builtin_amdgcn_wmma_f32_16x16x32_bf16(false, a, false, b0,
                                                       (short)0, acc0, false, false);
        acc1 = __builtin_amdgcn_wmma_f32_16x16x32_bf16(false, a, false, b1,
                                                       (short)0, acc1, false, false);
    }

    if (live) {
        int mh = (lane >> 4) << 3;       // D row offset
        float bias0 = fc_b[m], bias1 = fc_b[m + 16];
#pragma unroll
        for (int r = 0; r < 8; ++r) {
            size_t node = (size_t)(base + r + mh);
            hout[node * EMB + m]      = acc0[r] + bias0;
            hout[node * EMB + m + 16] = acc1[r] + bias1;
        }
    }
}

// ---------------------------------------------------------------------------
// Edge scatter-sum: one thread per (edge, feature). agg[dst] += h[src].
// Feature table + agg buffer are L2-resident (12.8 MB on 192 MB L2).
// ---------------------------------------------------------------------------
__global__ void agg_kernel(const float* __restrict__ h,
                           const int* __restrict__ src,
                           const int* __restrict__ dst,
                           float* __restrict__ agg,
                           int nEdges, int F, int logF)
{
    long long gid   = (long long)blockIdx.x * blockDim.x + threadIdx.x;
    long long total = (long long)nEdges << logF;
    if (gid >= total) return;
    int e = (int)(gid >> logF);
    int f = (int)(gid & (F - 1));
    int s = src[e], d = dst[e];
    atomicAdd(&agg[(size_t)d * F + f], h[(size_t)s * F + f]);
}

// ---------------------------------------------------------------------------
// GIN MLP: z = h + agg ; out = [ReLU](ReLU(z @ w1.T + b1) @ w2.T + b2)
// Fin in {32,16} (K padded to 32). Two chained WMMAs per 16-node tile.
// ---------------------------------------------------------------------------
__global__ void __launch_bounds__(256) gin_mlp_kernel(
    const float* __restrict__ hin, const float* __restrict__ agg,
    const __bf16* __restrict__ w1b, const float* __restrict__ b1,
    const __bf16* __restrict__ w2b, const float* __restrict__ b2,
    float* __restrict__ hout, int Fin, int relu_out)
{
    __shared__ __bf16 sT[WPB][16][16];   // mid-layer redistribution tile

    int t    = threadIdx.x;
    int wave = t >> 5;
    int lane = t & 31;
    int tile = blockIdx.x * WPB + wave;
    int base = tile * 16;
    bool live = (tile < N_TILES);

    int m  = lane & 15;
    int kA = (lane >> 4) << 3;
    int kB = (lane >> 4) << 4;
    int mh = (lane >> 4) << 3;

    // A fragment: z[m][k] = hin + agg, K padded to 32 with zeros
    v16bf a;
#pragma unroll
    for (int e = 0; e < 16; ++e) a[e] = (__bf16)0.0f;
    if (live) {
        const float* hp = hin + (size_t)(base + m) * Fin;
        const float* ap = agg + (size_t)(base + m) * Fin;
#pragma unroll
        for (int e = 0; e < 16; ++e) {
            int k = kB + e;
            if (k < Fin) a[e] = (__bf16)(hp[k] + ap[k]);
        }
    }

    // B1: w1.T as [n][k] bf16, stride 32
    v16bf bw1;
    const __bf16* bp1 = w1b + m * 32 + kB;
#pragma unroll
    for (int e = 0; e < 16; ++e) bw1[e] = bp1[e];

    v8f c1 = {};
    c1 = __builtin_amdgcn_wmma_f32_16x16x32_bf16(false, a, false, bw1,
                                                 (short)0, c1, false, false);

    // bias + ReLU, stage transposed into LDS for second-GEMM A layout
    float bias1 = b1[m];
#pragma unroll
    for (int r = 0; r < 8; ++r) {
        float v = c1[r] + bias1;
        sT[wave][r + mh][m] = (__bf16)(v > 0.0f ? v : 0.0f);
    }
    __syncthreads();

    v16bf a2;
#pragma unroll
    for (int e = 0; e < 16; ++e) {
        int k = kB + e;
        a2[e] = (k < HID) ? sT[wave][m][k] : (__bf16)0.0f;
    }
    v16bf bw2;
    const __bf16* bp2 = w2b + m * 32 + kB;
#pragma unroll
    for (int e = 0; e < 16; ++e) bw2[e] = bp2[e];

    v8f c2 = {};
    c2 = __builtin_amdgcn_wmma_f32_16x16x32_bf16(false, a2, false, bw2,
                                                 (short)0, c2, false, false);

    if (live) {
        float bias2 = b2[m];
#pragma unroll
        for (int r = 0; r < 8; ++r) {
            float v = c2[r] + bias2;
            if (relu_out) v = v > 0.0f ? v : 0.0f;
            hout[(size_t)(base + r + mh) * HID + m] = v;
        }
    }
}

// ---------------------------------------------------------------------------
// Readout: out[n] = dot(h[n], ro_w) + ro_b
// ---------------------------------------------------------------------------
__global__ void readout_kernel(const float* __restrict__ h,
                               const float* __restrict__ w,
                               const float* __restrict__ b,
                               float* __restrict__ out)
{
    int i = blockIdx.x * blockDim.x + threadIdx.x;
    if (i >= N_NODES) return;
    float acc = b[0];
    const float* hp = h + (size_t)i * HID;
#pragma unroll
    for (int k = 0; k < HID; ++k) acc += hp[k] * w[k];
    out[i] = acc;
}

// ---------------------------------------------------------------------------
extern "C" void kernel_launch(void* const* d_in, const int* in_sizes, int n_in,
                              void* d_out, int out_size, void* d_ws, size_t ws_size,
                              hipStream_t stream) {
    const float* x      = (const float*)d_in[0];
    const int*   ei     = (const int*)d_in[1];
    const float* conv_w = (const float*)d_in[2];
    const float* conv_b = (const float*)d_in[3];
    const float* fc_w   = (const float*)d_in[4];
    const float* fc_b   = (const float*)d_in[5];
    const float* g1w1   = (const float*)d_in[6];
    const float* g1b1   = (const float*)d_in[7];
    const float* g1w2   = (const float*)d_in[8];
    const float* g1b2   = (const float*)d_in[9];
    const float* g2w1   = (const float*)d_in[10];
    const float* g2b1   = (const float*)d_in[11];
    const float* g2w2   = (const float*)d_in[12];
    const float* g2b2   = (const float*)d_in[13];
    const float* ro_w   = (const float*)d_in[14];
    const float* ro_b   = (const float*)d_in[15];

    int nEdges = in_sizes[1] / 2;
    const int* src = ei;            // edge_index[0]
    const int* dst = ei + nEdges;   // edge_index[1]

    // workspace layout (all offsets 256B-aligned)
    char* ws = (char*)d_ws;
    size_t off = 0;
    float* h_emb  = (float*)(ws + off); off += (size_t)N_NODES * EMB * 4;   // 12.8 MB
    float* aggb   = (float*)(ws + off); off += (size_t)N_NODES * EMB * 4;   // 12.8 MB
    float* h_hid  = (float*)(ws + off); off += (size_t)N_NODES * HID * 4;   // 6.4 MB
    float* h_hid2 = (float*)(ws + off); off += (size_t)N_NODES * HID * 4;   // 6.4 MB
    __bf16* fcT = (__bf16*)(ws + off);  off += (size_t)EMB * K_PAD * 2;
    __bf16* w11 = (__bf16*)(ws + off);  off += (size_t)HID * 32 * 2;
    __bf16* w12 = (__bf16*)(ws + off);  off += (size_t)HID * 32 * 2;
    __bf16* w21 = (__bf16*)(ws + off);  off += (size_t)HID * 32 * 2;
    __bf16* w22 = (__bf16*)(ws + off);  off += (size_t)HID * 32 * 2;

    int tileBlocks = (N_TILES + WPB - 1) / WPB;   // 782

    prep_weights_kernel<<<40, 256, 0, stream>>>(fc_w, g1w1, g1w2, g2w1, g2w2,
                                                fcT, w11, w12, w21, w22);

    conv_fc_kernel<<<tileBlocks, 256, 0, stream>>>(x, conv_w, conv_b, fcT, fc_b, h_emb);

    // GIN layer 1
    hipMemsetAsync(aggb, 0, (size_t)N_NODES * EMB * 4, stream);
    {
        long long tot = (long long)nEdges * EMB;
        int blocks = (int)((tot + 255) / 256);
        agg_kernel<<<blocks, 256, 0, stream>>>(h_emb, src, dst, aggb, nEdges, EMB, 5);
    }
    gin_mlp_kernel<<<tileBlocks, 256, 0, stream>>>(h_emb, aggb, w11, g1b1, w12, g1b2,
                                                   h_hid, EMB, /*relu_out=*/1);

    // GIN layer 2
    hipMemsetAsync(aggb, 0, (size_t)N_NODES * HID * 4, stream);
    {
        long long tot = (long long)nEdges * HID;
        int blocks = (int)((tot + 255) / 256);
        agg_kernel<<<blocks, 256, 0, stream>>>(h_hid, src, dst, aggb, nEdges, HID, 4);
    }
    gin_mlp_kernel<<<tileBlocks, 256, 0, stream>>>(h_hid, aggb, w21, g2b1, w22, g2b2,
                                                   h_hid2, HID, /*relu_out=*/0);

    readout_kernel<<<(N_NODES + 255) / 256, 256, 0, stream>>>(h_hid2, ro_w, ro_b,
                                                              (float*)d_out);
}
